// Attention_61478161874995
// MI455X (gfx1250) — matrix operand order repaired and checked by reading, not compiled, not run on hardware
//
#include <hip/hip_runtime.h>

#define BB 8
#define SS 2048
#define DD 1024
#define MM (BB * SS) // 16384 rows

typedef __attribute__((ext_vector_type(16))) __bf16 v16bf;
typedef __attribute__((ext_vector_type(8)))  float  v8f;
typedef __attribute__((ext_vector_type(4)))  int    i4;
typedef __attribute__((ext_vector_type(4)))  float  f4;

union Frag { i4 i[2]; unsigned short us[16]; v16bf v; };
union Acc  { v8f v; float f[8]; };
union Pack8 { i4 i; unsigned short us[8]; };

__device__ __forceinline__ unsigned short f2bf(float f) {
  unsigned u = __builtin_bit_cast(unsigned, f);
  u += 0x7FFFu + ((u >> 16) & 1u); // round-to-nearest-even
  return (unsigned short)(u >> 16);
}

// ---------------------------------------------------------------------------
// Xb = bf16(X): one-time elementwise convert (8 elements / thread)
// ---------------------------------------------------------------------------
__global__ __launch_bounds__(256) void xconv_kernel(const float* __restrict__ X,
                                                    unsigned short* __restrict__ Xb) {
  const size_t idx = ((size_t)blockIdx.x * 256 + threadIdx.x) * 8;
  const f4* src = (const f4*)(X + idx);
  f4 f0 = src[0], f1 = src[1];
  Pack8 p;
  p.us[0] = f2bf(f0.x); p.us[1] = f2bf(f0.y); p.us[2] = f2bf(f0.z); p.us[3] = f2bf(f0.w);
  p.us[4] = f2bf(f1.x); p.us[5] = f2bf(f1.y); p.us[6] = f2bf(f1.z); p.us[7] = f2bf(f1.w);
  *(i4*)(Xb + idx) = p.i;
}

// ---------------------------------------------------------------------------
// Wt[n][k] = bf16(W[k][n])  (transpose so B-fragments are contiguous rows)
// ---------------------------------------------------------------------------
__global__ __launch_bounds__(256) void wconv_kernel(const float* __restrict__ W,
                                                    unsigned short* __restrict__ Wt) {
  int idx = blockIdx.x * 256 + threadIdx.x; // over D*D
  int k = idx >> 10, n = idx & (DD - 1);
  Wt[(size_t)n * DD + k] = f2bf(W[idx]);
}

// ---------------------------------------------------------------------------
// out = bf16((Xb @ W + b) * scale), one 32x64 tile per wave (8 WMMA / k-chunk).
// Xb: bf16 [M, D] row-major.  Wt: bf16 [N=D][K=D] (W transposed).
// transpose_out==0 : out bf16 [M, D] row-major   (Q, K)
// transpose_out==1 : out bf16 [B][D][S]          (V transposed for PV WMMA)
// ---------------------------------------------------------------------------
__global__ __launch_bounds__(256) void proj_kernel(const unsigned short* __restrict__ Xb,
                                                   const unsigned short* __restrict__ Wt,
                                                   const float* __restrict__ bias,
                                                   unsigned short* __restrict__ out,
                                                   int transpose_out, float out_scale) {
  const int lane = threadIdx.x & 31, wave = threadIdx.x >> 5;
  const int grp = lane >> 4, ln = lane & 15;
  const int m0 = (blockIdx.y * 8 + wave) * 32; // 512 m-tiles of 32 rows
  const int n0 = blockIdx.x * 64;              // 16  n-tiles of 64 cols

  Acc acc[2][4];
  for (int i = 0; i < 2; ++i)
    for (int j = 0; j < 4; ++j)
      for (int v = 0; v < 8; ++v) acc[i][j].f[v] = 0.f;

  const unsigned short* xrow0 = Xb + (size_t)(m0 + ln) * DD;
  const unsigned short* xrow1 = Xb + (size_t)(m0 + 16 + ln) * DD;
  const unsigned short* wrow0 = Wt + (size_t)(n0 + ln) * DD;

  for (int kc = 0; kc < DD / 32; ++kc) {
    const int kb = kc * 32;
    Frag a0, a1;
    { const i4* p = (const i4*)(xrow0 + kb + grp * 16); a0.i[0] = p[0]; a0.i[1] = p[1]; }
    { const i4* p = (const i4*)(xrow1 + kb + grp * 16); a1.i[0] = p[0]; a1.i[1] = p[1]; }
    for (int j = 0; j < 4; ++j) {
      const i4* pb = (const i4*)(wrow0 + (size_t)j * 16 * DD + kb + grp * 16);
      Frag bf; bf.i[0] = pb[0]; bf.i[1] = pb[1];
      acc[0][j].v = __builtin_amdgcn_wmma_f32_16x16x32_bf16(false, a0.v, false, bf.v,
                                                            (short)0, acc[0][j].v, false, false);
      acc[1][j].v = __builtin_amdgcn_wmma_f32_16x16x32_bf16(false, a1.v, false, bf.v,
                                                            (short)0, acc[1][j].v, false, false);
    }
  }

  if (!transpose_out) {
    for (int j = 0; j < 4; ++j) {
      const float bv = bias[n0 + j * 16 + ln];
      for (int i = 0; i < 2; ++i)
        for (int v = 0; v < 8; ++v)
          out[(size_t)(m0 + i * 16 + grp * 8 + v) * DD + (n0 + j * 16 + ln)] =
              f2bf((acc[i][j].f[v] + bv) * out_scale);
    }
  } else {
    // rows of one m-subtile lie in one batch (32 | S); lane's 8 values per
    // (i,j) are 8 consecutive s positions -> single 16-byte store
    const int bidx = m0 / SS;
    for (int j = 0; j < 4; ++j) {
      const float bv = bias[n0 + j * 16 + ln];
      for (int i = 0; i < 2; ++i) {
        const int s0 = (m0 % SS) + i * 16 + grp * 8;
        Pack8 st;
        for (int v = 0; v < 8; ++v) st.us[v] = f2bf((acc[i][j].f[v] + bv) * out_scale);
        *(i4*)(out + (size_t)bidx * DD * SS + (size_t)(n0 + j * 16 + ln) * SS + s0) = st.i;
      }
    }
  }
}

// ---------------------------------------------------------------------------
// Flash attention: 1 block = 8 waves = one 16-query tile of one batch.
// Waves split key blocks (32 keys each inside a 256-key block) for scores,
// and split the 1024-dim output into 128-dim slices for PV.
// ---------------------------------------------------------------------------
__global__ __launch_bounds__(256) void attn_kernel(const unsigned short* __restrict__ Qb,
                                                   const unsigned short* __restrict__ Kb,
                                                   const unsigned short* __restrict__ Vt,
                                                   float* __restrict__ out) {
  __shared__ __align__(16) unsigned short Qs[16 * DD];  // 32 KB
  __shared__ __align__(16) unsigned short Ps[16 * 256]; // 8 KB
  __shared__ float redm[8][16];
  __shared__ float reds[8][16];

  const int t = threadIdx.x;
  const int lane = t & 31, wave = t >> 5;
  const int grp = lane >> 4, ln = lane & 15;
  const int bidx = blockIdx.x >> 7;            // 128 query tiles per batch
  const int q0 = (blockIdx.x & 127) * 16;

  { // stage the 16 x D bf16 Q tile (rows are contiguous in Qb)
    const i4* src = (const i4*)(Qb + ((size_t)bidx * SS + q0) * DD);
    i4* dst = (i4*)Qs;
    for (int i = t; i < 16 * DD / 8; i += 256) dst[i] = src[i];
  }
  __syncthreads();

  float m_run[8], l_run[8];
  Acc oa[8];
  for (int v = 0; v < 8; ++v) {
    m_run[v] = -1e30f; l_run[v] = 0.f;
    for (int j = 0; j < 8; ++j) oa[v].f[j] = 0.f;
  }

  const int dbase = wave * 128;
  const size_t kbase = (size_t)bidx * SS * DD; // Kb batch base
  const size_t vbase = (size_t)bidx * DD * SS; // Vt batch base
  const int nkb = q0 / 256 + 1;                // causal: key blocks needed

  for (int kb = 0; kb < nkb; ++kb) {
    const int kblk = kb * 256;
    const int kw = kblk + wave * 32; // this wave's 32 keys

    // ---- scores: S[16 x 32] = Q Ktile^T (scale folded into Qb) ----
    Acc s0, s1;
    for (int v = 0; v < 8; ++v) { s0.f[v] = 0.f; s1.f[v] = 0.f; }
    for (int kc = 0; kc < 32; ++kc) {
      Frag a;
      const i4* pa = (const i4*)(Qs + ln * DD + kc * 32 + grp * 16);
      a.i[0] = pa[0]; a.i[1] = pa[1];
      const i4* pk0 = (const i4*)(Kb + kbase + (size_t)(kw + ln) * DD + kc * 32 + grp * 16);
      const i4* pk1 = (const i4*)(Kb + kbase + (size_t)(kw + 16 + ln) * DD + kc * 32 + grp * 16);
      Frag b0, b1;
      b0.i[0] = pk0[0]; b0.i[1] = pk0[1];
      b1.i[0] = pk1[0]; b1.i[1] = pk1[1];
      s0.v = __builtin_amdgcn_wmma_f32_16x16x32_bf16(false, a.v, false, b0.v,
                                                     (short)0, s0.v, false, false);
      s1.v = __builtin_amdgcn_wmma_f32_16x16x32_bf16(false, a.v, false, b1.v,
                                                     (short)0, s1.v, false, false);
    }
    // ---- causal mask (analytic; C layout: row = grp*8+v, col = ln) ----
    for (int v = 0; v < 8; ++v) {
      const int qrow = q0 + grp * 8 + v;
      if (kw + ln > qrow)      s0.f[v] = -1e30f;
      if (kw + 16 + ln > qrow) s1.f[v] = -1e30f;
    }
    // ---- row max: butterfly within 16-lane group, then cross-wave via LDS ----
    float mloc[8];
    for (int v = 0; v < 8; ++v) mloc[v] = fmaxf(s0.f[v], s1.f[v]);
    for (int off = 1; off < 16; off <<= 1)
      for (int v = 0; v < 8; ++v)
        mloc[v] = fmaxf(mloc[v], __shfl_xor(mloc[v], off, 32));
    if (ln == 0)
      for (int v = 0; v < 8; ++v) redm[wave][grp * 8 + v] = mloc[v];
    __syncthreads();

    float m_new[8], alpha[8];
    for (int v = 0; v < 8; ++v) {
      float bm = -1e30f;
      for (int w = 0; w < 8; ++w) bm = fmaxf(bm, redm[w][grp * 8 + v]);
      m_new[v] = fmaxf(m_run[v], bm);
      alpha[v] = __expf(m_run[v] - m_new[v]);
    }
    // ---- p = exp(s - m_new); stash bf16 P to LDS; local row sums ----
    float sl[8];
    for (int v = 0; v < 8; ++v) {
      const float p0 = __expf(s0.f[v] - m_new[v]);
      const float p1 = __expf(s1.f[v] - m_new[v]);
      sl[v] = p0 + p1;
      Ps[(grp * 8 + v) * 256 + wave * 32 + ln]      = f2bf(p0);
      Ps[(grp * 8 + v) * 256 + wave * 32 + 16 + ln] = f2bf(p1);
    }
    for (int off = 1; off < 16; off <<= 1)
      for (int v = 0; v < 8; ++v) sl[v] += __shfl_xor(sl[v], off, 32);
    if (ln == 0)
      for (int v = 0; v < 8; ++v) reds[wave][grp * 8 + v] = sl[v];
    __syncthreads();

    for (int v = 0; v < 8; ++v) {
      float bs = 0.f;
      for (int w = 0; w < 8; ++w) bs += reds[w][grp * 8 + v];
      l_run[v] = l_run[v] * alpha[v] + bs;
      m_run[v] = m_new[v];
    }
    for (int dc = 0; dc < 8; ++dc)
      for (int v = 0; v < 8; ++v) oa[dc].f[v] *= alpha[v];

    // ---- O += P @ Vtile for this wave's 128-dim slice ----
    for (int kc = 0; kc < 8; ++kc) {
      Frag a;
      const i4* pa = (const i4*)(Ps + ln * 256 + kc * 32 + grp * 16);
      a.i[0] = pa[0]; a.i[1] = pa[1];
      for (int dc = 0; dc < 8; ++dc) {
        const int dim = dbase + dc * 16 + ln;
        const i4* pv = (const i4*)(Vt + vbase + (size_t)dim * SS + kblk + kc * 32 + grp * 16);
        Frag bf; bf.i[0] = pv[0]; bf.i[1] = pv[1];
        oa[dc].v = __builtin_amdgcn_wmma_f32_16x16x32_bf16(false, a.v, false, bf.v,
                                                           (short)0, oa[dc].v, false, false);
      }
    }
    __syncthreads(); // Ps/redm reused next key block
  }

  // ---- epilogue: multiply by reciprocal row sums, store fp32 ----
  float inv[8];
  for (int v = 0; v < 8; ++v) inv[v] = 1.0f / l_run[v];
  for (int dc = 0; dc < 8; ++dc)
    for (int v = 0; v < 8; ++v) {
      const float r = oa[dc].f[v] * inv[v];
      out[((size_t)bidx * SS + q0 + grp * 8 + v) * DD + dbase + dc * 16 + ln] = r;
    }
}

// ---------------------------------------------------------------------------
extern "C" void kernel_launch(void* const* d_in, const int* in_sizes, int n_in,
                              void* d_out, int out_size, void* d_ws, size_t ws_size,
                              hipStream_t stream) {
  (void)in_sizes; (void)n_in; (void)out_size; (void)ws_size;
  const float* q  = (const float*)d_in[0];
  const float* k  = (const float*)d_in[1];
  const float* vv = (const float*)d_in[2];
  // d_in[3] = causal mask (bool) -- handled analytically in attn_kernel
  const float* Wq = (const float*)d_in[4];
  const float* bq = (const float*)d_in[5];
  const float* Wk = (const float*)d_in[6];
  const float* bk = (const float*)d_in[7];
  const float* Wv = (const float*)d_in[8];
  const float* bv = (const float*)d_in[9];
  float* out = (float*)d_out;

  unsigned short* ws = (unsigned short*)d_ws;
  const size_t nBSD = (size_t)BB * SS * DD;
  unsigned short* Qb  = ws;                       // bf16 [B,S,D], pre-scaled 1/sqrt(D)
  unsigned short* Kbf = Qb  + nBSD;               // bf16 [B,S,D]
  unsigned short* Vt  = Kbf + nBSD;               // bf16 [B,D,S]
  unsigned short* Xb  = Vt  + nBSD;               // bf16 staging, reused 3x
  unsigned short* Wqt = Xb  + nBSD;               // bf16 [D,D] transposed
  unsigned short* Wkt = Wqt + (size_t)DD * DD;
  unsigned short* Wvt = Wkt + (size_t)DD * DD;

  dim3 blk(256);
  wconv_kernel<<<DD * DD / 256, blk, 0, stream>>>(Wq, Wqt);
  wconv_kernel<<<DD * DD / 256, blk, 0, stream>>>(Wk, Wkt);
  wconv_kernel<<<DD * DD / 256, blk, 0, stream>>>(Wv, Wvt);

  const int xgrid = (int)(nBSD / (256 * 8));
  dim3 pgrid(DD / 64, MM / 32 / 8);

  // stream-ordered: convert each input into the shared staging buffer, then GEMM
  xconv_kernel<<<xgrid, blk, 0, stream>>>(q, Xb);
  proj_kernel<<<pgrid, blk, 0, stream>>>(Xb, Wqt, bq, Qb, 0, 0.03125f); // 1/sqrt(1024)
  xconv_kernel<<<xgrid, blk, 0, stream>>>(k, Xb);
  proj_kernel<<<pgrid, blk, 0, stream>>>(Xb, Wkt, bk, Kbf, 0, 1.0f);
  xconv_kernel<<<xgrid, blk, 0, stream>>>(vv, Xb);
  proj_kernel<<<pgrid, blk, 0, stream>>>(Xb, Wvt, bv, Vt, 1, 1.0f);

  attn_kernel<<<BB * SS / 16, blk, 0, stream>>>(Qb, Kbf, Vt, out);
}
